// EncoderBlock_1735166787937
// MI455X (gfx1250) — matrix operand order repaired and checked by reading, not compiled, not run on hardware
//
#include <hip/hip_runtime.h>

// ---------------------------------------------------------------------------
// Types
// ---------------------------------------------------------------------------
typedef __attribute__((ext_vector_type(16))) __bf16 v16bf;
typedef __attribute__((ext_vector_type(8)))  __bf16 v8bf;
typedef __attribute__((ext_vector_type(8)))  float  v8f;
typedef __attribute__((ext_vector_type(4)))  unsigned int v4u;

// Problem constants
#define BATCH 2
#define SEQ   2048
#define DMODEL 768
#define HEADS 12
#define HDIM  64
#define FFN   3072
#define MTOT  (BATCH * SEQ)   // 4096

// ---------------------------------------------------------------------------
// Helpers
// ---------------------------------------------------------------------------
static __device__ __forceinline__ unsigned short f2bfu(float f) {
  union { float f; unsigned int u; } c; c.f = f;
  unsigned int u = c.u;
  unsigned int r = (u + 0x7FFFu + ((u >> 16) & 1u)) >> 16;
  return (unsigned short)r;
}

// Build a 16-element bf16 fragment from two contiguous 8-element (16B) runs.
static __device__ __forceinline__ v16bf ldfrag(const unsigned short* p0,
                                               const unsigned short* p1) {
  v8bf a = *reinterpret_cast<const v8bf*>(p0);
  v8bf b = *reinterpret_cast<const v8bf*>(p1);
  return __builtin_shufflevector(a, b, 0,1,2,3,4,5,6,7,8,9,10,11,12,13,14,15);
}

static __device__ __forceinline__ v8f wmma_bf16(v16bf a, v16bf b, v8f c) {
  return __builtin_amdgcn_wmma_f32_16x16x32_bf16(
      /*neg_a=*/false, a, /*neg_b=*/false, b,
      /*c_mod=*/(short)0, c, /*reuse_a=*/false, /*reuse_b=*/false);
}

// ---- DPP lane-xor permutes within a 16-lane DPP row (pure VALU, no LDS) ----
static __device__ __forceinline__ float dpp_xor1(float x) {
  int i = __float_as_int(x);
  // quad_perm [1,0,3,2]
  int r = __builtin_amdgcn_update_dpp(i, i, 0xB1, 0xF, 0xF, false);
  return __int_as_float(r);
}
static __device__ __forceinline__ float dpp_xor2(float x) {
  int i = __float_as_int(x);
  // quad_perm [2,3,0,1]
  int r = __builtin_amdgcn_update_dpp(i, i, 0x4E, 0xF, 0xF, false);
  return __int_as_float(r);
}
static __device__ __forceinline__ float dpp_xor4(float x) {
  int i = __float_as_int(x);
  // dpp8: lane i <- lane (i^4) within each group of 8
  int r = __builtin_amdgcn_mov_dpp8(
      i, (4 | (5 << 3) | (6 << 6) | (7 << 9) |
          (0 << 12) | (1 << 15) | (2 << 18) | (3 << 21)));
  return __int_as_float(r);
}
static __device__ __forceinline__ float dpp_xor8(float x) {
  int i = __float_as_int(x);
  // row_ror:8 == lane i <- lane (i^8) within each 16-lane row
  int r = __builtin_amdgcn_update_dpp(i, i, 0x128, 0xF, 0xF, false);
  return __int_as_float(r);
}
static __device__ __forceinline__ float rowmax16(float x) {
  x = fmaxf(x, dpp_xor1(x));
  x = fmaxf(x, dpp_xor2(x));
  x = fmaxf(x, dpp_xor4(x));
  x = fmaxf(x, dpp_xor8(x));
  return x;
}
static __device__ __forceinline__ float rowsum16(float x) {
  x += dpp_xor1(x);
  x += dpp_xor2(x);
  x += dpp_xor4(x);
  x += dpp_xor8(x);
  return x;
}

// ---------------------------------------------------------------------------
// f32 -> bf16 convert (grid-stride)
// ---------------------------------------------------------------------------
__global__ __launch_bounds__(256) void cvt_bf16_kernel(
    const float* __restrict__ in, unsigned short* __restrict__ out, int n) {
  for (int i = blockIdx.x * blockDim.x + threadIdx.x; i < n;
       i += gridDim.x * blockDim.x)
    out[i] = f2bfu(in[i]);
}

// ---------------------------------------------------------------------------
// Generic WMMA GEMM:  C[M,N] = A[M,K] (bf16) @ W[N,K]^T (bf16) + bias
// Block = 256 threads (8 waves). Block tile 128(M) x 128(N), K-step 32.
// Wave (wm in 0..3, wn in 0..1) owns a 32x64 output strip (2x4 WMMA tiles).
// MODE 0: f32 out row-major [M,N]
// MODE 1: bf16 out scattered to [b, h, seq, 64]      (Q, K)
// MODE 2: bf16 out scattered to [b, h, 64, seq]      (V, pre-transposed)
// MODE 3: exact-GELU then bf16 out row-major [M,N]
// ---------------------------------------------------------------------------
template <int MODE>
__global__ __launch_bounds__(256) void gemm_bf16_kernel(
    const unsigned short* __restrict__ A,   // [M,K] bf16
    const unsigned short* __restrict__ W,   // [N,K] bf16
    const float* __restrict__ bias,         // [N]
    void* __restrict__ out, int M, int N, int K) {
  __shared__ unsigned short As[128 * 32];
  __shared__ unsigned short Bs[128 * 32];

  const int t  = threadIdx.x;
  const int l  = t & 31;
  const int w  = t >> 5;
  const int wm = w & 3;          // 4 M-positions of 32 rows
  const int wn = w >> 2;         // 2 N-positions of 64 cols
  const int ln = l & 15;
  const int hi = l >> 4;
  const int m0 = blockIdx.x * 128;
  const int n0 = blockIdx.y * 128;

  v8f acc[2][4] = {};

  for (int k0 = 0; k0 < K; k0 += 32) {
    __syncthreads();
    // Stage A and B tiles: 128 rows x 32 cols bf16 each; 16B chunks.
#pragma unroll
    for (int i = 0; i < 2; ++i) {
      int idx = t + i * 256;
      int m = idx >> 2, seg = idx & 3;
      *(v4u*)&As[m * 32 + seg * 8] =
          *(const v4u*)&A[(size_t)(m0 + m) * K + k0 + seg * 8];
      *(v4u*)&Bs[m * 32 + seg * 8] =
          *(const v4u*)&W[(size_t)(n0 + m) * K + k0 + seg * 8];
    }
    if (k0 + 32 < K) {
      // Prefetch next K-tiles into cache (global_prefetch_b8).
      __builtin_prefetch(&A[(size_t)(m0 + (t >> 1)) * K + k0 + 32], 0, 0);
      __builtin_prefetch(&W[(size_t)(n0 + (t >> 1)) * K + k0 + 32], 0, 0);
    }
    __syncthreads();

    v16bf af[2];
#pragma unroll
    for (int mi = 0; mi < 2; ++mi) {
      const unsigned short* ap = &As[(wm * 32 + mi * 16 + ln) * 32];
      af[mi] = ldfrag(ap + hi * 8, ap + 16 + hi * 8);
    }
#pragma unroll
    for (int j = 0; j < 4; ++j) {
      const unsigned short* bp = &Bs[(wn * 64 + j * 16 + ln) * 32];
      v16bf bf = ldfrag(bp + hi * 8, bp + 16 + hi * 8);
#pragma unroll
      for (int mi = 0; mi < 2; ++mi)
        acc[mi][j] = wmma_bf16(af[mi], bf, acc[mi][j]);
    }
  }

  // Epilogue. C layout: row = 8*hi + r, col = ln (within 16x16 tile).
#pragma unroll
  for (int mi = 0; mi < 2; ++mi) {
#pragma unroll
    for (int j = 0; j < 4; ++j) {
#pragma unroll
      for (int r = 0; r < 8; ++r) {
        int gm = m0 + wm * 32 + mi * 16 + 8 * hi + r;
        int gn = n0 + wn * 64 + j * 16 + ln;
        float v = acc[mi][j][r] + bias[gn];
        if (MODE == 0) {
          ((float*)out)[(size_t)gm * N + gn] = v;
        } else if (MODE == 1) {
          int bi = gm >> 11, q = gm & (SEQ - 1);
          int h = gn >> 6, d = gn & (HDIM - 1);
          ((unsigned short*)out)[(((size_t)(bi * HEADS + h) * SEQ + q) * HDIM) + d] =
              f2bfu(v);
        } else if (MODE == 2) {
          int bi = gm >> 11, q = gm & (SEQ - 1);
          int h = gn >> 6, d = gn & (HDIM - 1);
          ((unsigned short*)out)[(((size_t)(bi * HEADS + h) * HDIM + d) * SEQ) + q] =
              f2bfu(v);
        } else if (MODE == 3) {
          float g = 0.5f * v * (1.0f + erff(v * 0.7071067811865475f));
          ((unsigned short*)out)[(size_t)gm * N + gn] = f2bfu(g);
        }
      }
    }
  }
}

// ---------------------------------------------------------------------------
// Flash attention (barrier-free):
//   Q,K bf16 [b,h,seq,64], Vt bf16 [b,h,64,seq] -> O bf16 [b, seq, h*64]
// Grid: (SEQ/128, BATCH*HEADS). Block 256 = 8 waves; wave owns 16 queries.
// Per 32-key chunk: 4 score WMMAs (K frags from global), DPP softmax,
// P round-trip through wave-private LDS, 4 PV WMMAs (Vt frags from global).
// ---------------------------------------------------------------------------
__global__ __launch_bounds__(256) void flash_attn_kernel(
    const unsigned short* __restrict__ Q, const unsigned short* __restrict__ Kk,
    const unsigned short* __restrict__ Vt, unsigned short* __restrict__ O) {
  __shared__ unsigned short Ps[8][16 * 32]; // per-wave P tile 16x32

  const int t  = threadIdx.x;
  const int l  = t & 31;
  const int w  = t >> 5;
  const int ln = l & 15;
  const int hi = l >> 4;
  const int bh = blockIdx.y;          // bi*HEADS + h
  const int bi = bh / HEADS;
  const int h  = bh - bi * HEADS;
  const int q0 = blockIdx.x * 128 + w * 16;

  // Q fragments for the 16 queries this wave owns (K-dim 64 = 2 steps).
  const unsigned short* qb = Q + ((size_t)bh * SEQ + q0 + ln) * HDIM;
  v16bf qf0 = ldfrag(qb + hi * 8,      qb + 16 + hi * 8);
  v16bf qf1 = ldfrag(qb + 32 + hi * 8, qb + 48 + hi * 8);

  float mrow[8], lrow[8];
  v8f oacc[4] = {};
#pragma unroll
  for (int r = 0; r < 8; ++r) { mrow[r] = -3.0e38f; lrow[r] = 0.0f; }

  for (int key0 = 0; key0 < SEQ; key0 += 32) {
    // Scores: two 16-key subtiles, each accumulating over K=64 (2 steps).
    v8f sc[2] = {};
#pragma unroll
    for (int j2 = 0; j2 < 2; ++j2) {
      const unsigned short* kb =
          Kk + ((size_t)bh * SEQ + key0 + j2 * 16 + ln) * HDIM;
      v16bf kf0 = ldfrag(kb + hi * 8,      kb + 16 + hi * 8);
      v16bf kf1 = ldfrag(kb + 32 + hi * 8, kb + 48 + hi * 8);
      sc[j2] = wmma_bf16(qf0, kf0, sc[j2]);
      sc[j2] = wmma_bf16(qf1, kf1, sc[j2]);
    }

    // Online softmax (scale 1/sqrt(64) = 0.125), DPP reductions.
    float scl_r[8];
#pragma unroll
    for (int r = 0; r < 8; ++r) {
      float a0 = sc[0][r] * 0.125f;
      float a1 = sc[1][r] * 0.125f;
      float mx = rowmax16(fmaxf(a0, a1));
      float mnew = fmaxf(mrow[r], mx);
      float scl  = __expf(mrow[r] - mnew);
      mrow[r] = mnew;
      float p0 = __expf(a0 - mnew);
      float p1 = __expf(a1 - mnew);
      float rs = rowsum16(p0 + p1);
      lrow[r] = lrow[r] * scl + rs;
      scl_r[r] = scl;
      // Store P (C layout -> LDS row-major 16x32)
      Ps[w][(8 * hi + r) * 32 + ln]      = f2bfu(p0);
      Ps[w][(8 * hi + r) * 32 + 16 + ln] = f2bfu(p1);
    }
#pragma unroll
    for (int j = 0; j < 4; ++j)
#pragma unroll
      for (int r = 0; r < 8; ++r) oacc[j][r] *= scl_r[r];

    // Wave-private LDS RAW: DS ops are in-order per wave; wait for safety.
    asm volatile("s_wait_dscnt 0" ::: "memory");

    // Reload P as A-fragment (16x32) and do P @ V using pre-transposed Vt:
    // B[k,n] = V[key0+k][n] = Vt[bh][n][key0+k]  (key-contiguous rows).
    const unsigned short* pp = &Ps[w][ln * 32];
    v16bf pf = ldfrag(pp + hi * 8, pp + 16 + hi * 8);
#pragma unroll
    for (int j = 0; j < 4; ++j) {
      const unsigned short* vp =
          Vt + ((size_t)bh * HDIM + j * 16 + ln) * SEQ + key0;
      v16bf vf = ldfrag(vp + hi * 8, vp + 16 + hi * 8);
      oacc[j] = wmma_bf16(pf, vf, oacc[j]);
    }
  }

  // Write O[bi, q, h*64 + d] bf16 (row-major [MTOT, DMODEL]).
#pragma unroll
  for (int j = 0; j < 4; ++j)
#pragma unroll
    for (int r = 0; r < 8; ++r) {
      int gq = q0 + 8 * hi + r;
      float val = oacc[j][r] / lrow[r];
      O[((size_t)(bi * SEQ + gq)) * DMODEL + h * HDIM + j * 16 + ln] =
          f2bfu(val);
    }
}

// ---------------------------------------------------------------------------
// Residual add + LayerNorm over 768. One block per row. outbf may be null.
// ---------------------------------------------------------------------------
__global__ __launch_bounds__(256) void residual_ln_kernel(
    const float* __restrict__ xa, const float* __restrict__ xb,
    const float* __restrict__ g, const float* __restrict__ be,
    float* __restrict__ outf, unsigned short* __restrict__ outbf) {
  __shared__ float ws[8], wq[8];
  const int row = blockIdx.x;
  const int t = threadIdx.x;
  const float* pa = xa + (size_t)row * DMODEL;
  const float* pb = xb + (size_t)row * DMODEL;
  float v[3], s = 0.f, sq = 0.f;
#pragma unroll
  for (int i = 0; i < 3; ++i) {
    float x = pa[t + i * 256] + pb[t + i * 256];
    v[i] = x; s += x; sq += x * x;
  }
  // Wave reduce: DPP within 16, then one shfl across the two halves.
  s = rowsum16(s);   sq = rowsum16(sq);
  s += __shfl_xor(s, 16, 32);
  sq += __shfl_xor(sq, 16, 32);
  if ((t & 31) == 0) { ws[t >> 5] = s; wq[t >> 5] = sq; }
  __syncthreads();
  s = 0.f; sq = 0.f;
#pragma unroll
  for (int i = 0; i < 8; ++i) { s += ws[i]; sq += wq[i]; }
  float mu  = s * (1.0f / DMODEL);
  float var = sq * (1.0f / DMODEL) - mu * mu;
  float rstd = rsqrtf(var + 1e-5f);
#pragma unroll
  for (int i = 0; i < 3; ++i) {
    int c = t + i * 256;
    float y = (v[i] - mu) * rstd * g[c] + be[c];
    outf[(size_t)row * DMODEL + c] = y;
    if (outbf) outbf[(size_t)row * DMODEL + c] = f2bfu(y);
  }
}

// ---------------------------------------------------------------------------
// Launch
// ---------------------------------------------------------------------------
extern "C" void kernel_launch(void* const* d_in, const int* in_sizes, int n_in,
                              void* d_out, int out_size, void* d_ws,
                              size_t ws_size, hipStream_t stream) {
  (void)in_sizes; (void)n_in; (void)out_size; (void)ws_size;

  const float* x    = (const float*)d_in[0];
  const float* Wq   = (const float*)d_in[1];
  const float* bq   = (const float*)d_in[2];
  const float* Wk   = (const float*)d_in[3];
  const float* bk   = (const float*)d_in[4];
  const float* Wv   = (const float*)d_in[5];
  const float* bv   = (const float*)d_in[6];
  const float* Wo   = (const float*)d_in[7];
  const float* bo   = (const float*)d_in[8];
  const float* g1   = (const float*)d_in[9];
  const float* be1  = (const float*)d_in[10];
  const float* g2   = (const float*)d_in[11];
  const float* be2  = (const float*)d_in[12];
  const float* W1   = (const float*)d_in[13];
  const float* b1   = (const float*)d_in[14];
  const float* W2   = (const float*)d_in[15];
  const float* b2   = (const float*)d_in[16];

  // Workspace carve-up (256B aligned).
  char* ws = (char*)d_ws;
  size_t off = 0;
  auto carve = [&](size_t bytes) {
    char* p = ws + off;
    off += (bytes + 255) & ~(size_t)255;
    return p;
  };
  unsigned short* Xbf   = (unsigned short*)carve((size_t)MTOT * DMODEL * 2);
  unsigned short* Wqbf  = (unsigned short*)carve((size_t)DMODEL * DMODEL * 2);
  unsigned short* Wkbf  = (unsigned short*)carve((size_t)DMODEL * DMODEL * 2);
  unsigned short* Wvbf  = (unsigned short*)carve((size_t)DMODEL * DMODEL * 2);
  unsigned short* Wobf  = (unsigned short*)carve((size_t)DMODEL * DMODEL * 2);
  unsigned short* W1bf  = (unsigned short*)carve((size_t)FFN * DMODEL * 2);
  unsigned short* W2bf  = (unsigned short*)carve((size_t)DMODEL * FFN * 2);
  unsigned short* Qbf   = (unsigned short*)carve((size_t)MTOT * DMODEL * 2);
  unsigned short* Kbf   = (unsigned short*)carve((size_t)MTOT * DMODEL * 2);
  unsigned short* Vtbf  = (unsigned short*)carve((size_t)MTOT * DMODEL * 2);
  unsigned short* Attnbf= (unsigned short*)carve((size_t)MTOT * DMODEL * 2);
  float*          attnpj= (float*)carve((size_t)MTOT * DMODEL * 4);
  float*          x1f   = (float*)carve((size_t)MTOT * DMODEL * 4);
  unsigned short* x1bf  = (unsigned short*)carve((size_t)MTOT * DMODEL * 2);
  unsigned short* Hbf   = (unsigned short*)carve((size_t)MTOT * FFN * 2);
  float*          ffnpj = (float*)carve((size_t)MTOT * DMODEL * 4);

  // 1) Convert activations + weights to bf16.
  cvt_bf16_kernel<<<1024, 256, 0, stream>>>(x,  Xbf,  MTOT * DMODEL);
  cvt_bf16_kernel<<<512,  256, 0, stream>>>(Wq, Wqbf, DMODEL * DMODEL);
  cvt_bf16_kernel<<<512,  256, 0, stream>>>(Wk, Wkbf, DMODEL * DMODEL);
  cvt_bf16_kernel<<<512,  256, 0, stream>>>(Wv, Wvbf, DMODEL * DMODEL);
  cvt_bf16_kernel<<<512,  256, 0, stream>>>(Wo, Wobf, DMODEL * DMODEL);
  cvt_bf16_kernel<<<1024, 256, 0, stream>>>(W1, W1bf, FFN * DMODEL);
  cvt_bf16_kernel<<<1024, 256, 0, stream>>>(W2, W2bf, DMODEL * FFN);

  // 2) QKV projections. Q,K -> [b,h,seq,64]; V -> [b,h,64,seq] (transposed).
  dim3 gqkv(MTOT / 128, DMODEL / 128);
  gemm_bf16_kernel<1><<<gqkv, 256, 0, stream>>>(Xbf, Wqbf, bq, Qbf,
                                                MTOT, DMODEL, DMODEL);
  gemm_bf16_kernel<1><<<gqkv, 256, 0, stream>>>(Xbf, Wkbf, bk, Kbf,
                                                MTOT, DMODEL, DMODEL);
  gemm_bf16_kernel<2><<<gqkv, 256, 0, stream>>>(Xbf, Wvbf, bv, Vtbf,
                                                MTOT, DMODEL, DMODEL);

  // 3) Flash attention -> Attnbf [MTOT, DMODEL] bf16.
  flash_attn_kernel<<<dim3(SEQ / 128, BATCH * HEADS), 256, 0, stream>>>(
      Qbf, Kbf, Vtbf, Attnbf);

  // 4) Output projection (f32).
  gemm_bf16_kernel<0><<<gqkv, 256, 0, stream>>>(Attnbf, Wobf, bo, attnpj,
                                                MTOT, DMODEL, DMODEL);

  // 5) x1 = LN(x + attn) -> f32 + bf16.
  residual_ln_kernel<<<MTOT, 256, 0, stream>>>(x, attnpj, g1, be1, x1f, x1bf);

  // 6) FFN1 with exact GELU -> bf16 [MTOT, FFN].
  gemm_bf16_kernel<3><<<dim3(MTOT / 128, FFN / 128), 256, 0, stream>>>(
      x1bf, W1bf, b1, Hbf, MTOT, FFN, DMODEL);

  // 7) FFN2 -> f32.
  gemm_bf16_kernel<0><<<gqkv, 256, 0, stream>>>(Hbf, W2bf, b2, ffnpj,
                                                MTOT, DMODEL, FFN);

  // 8) out = LN(x1 + ffn).
  residual_ln_kernel<<<MTOT, 256, 0, stream>>>(x1f, ffnpj, g2, be2,
                                               (float*)d_out, nullptr);
}
